// MultiHeadAttention_4655744549359
// MI455X (gfx1250) — matrix operand order repaired and checked by reading, not compiled, not run on hardware
//
#include <hip/hip_runtime.h>

// ---------------------------------------------------------------------------
// MHA forward for B=2, S=2048, N_STATE=1024, H=16, Dh=64 on gfx1250 (CDNA5).
// bf16 WMMA (v_wmma_f32_16x16x32_bf16) everywhere; fp32 accumulate + softmax.
// Flash kernel: K/V tiles double-buffered in LDS via GLOBAL_LOAD_ASYNC_TO_LDS
// (ASYNCcnt), shared by the block's 8 waves; softmax row-sums via ones-matrix
// WMMA; row-max via DPP.
// ---------------------------------------------------------------------------

typedef __bf16 bf16;
typedef __attribute__((ext_vector_type(16))) __bf16 v16bf;
typedef __attribute__((ext_vector_type(8)))  __bf16 v8bf;
typedef __attribute__((ext_vector_type(4)))  __bf16 v4bf;
typedef __attribute__((ext_vector_type(8)))  float  v8f;

#define DEVINL __device__ __forceinline__

static constexpr int Bn     = 2;
static constexpr int S      = 2048;
static constexpr int NSTATE = 1024;
static constexpr int DHEAD  = 64;
static constexpr float QK_SCALE = 0.35355339059327373f; // 64^-0.25
static constexpr float NEG_BIG  = -1e9f;

// --- A fragment: 16x32 (MxK) bf16, source row-major (k contiguous), ld elems.
DEVINL v16bf load_frag_a(const bf16* base, int ld, int lane) {
  const int m = lane & 15, half = lane >> 4;
  const bf16* p0 = base + m * ld + half * 8;
  v8bf lo = *(const v8bf*)(p0);
  v8bf hi = *(const v8bf*)(p0 + 16);
  v16bf a;
#pragma unroll
  for (int i = 0; i < 8; ++i) { a[i] = lo[i]; a[i + 8] = hi[i]; }
  return a;
}

// --- B fragment: 32x16 (KxN) bf16, source stored N-major-K-contiguous.
DEVINL v16bf load_frag_b(const bf16* base, int ld, int lane) {
  const int n = lane & 15, half = lane >> 4;
  const bf16* p = base + n * ld + half * 8;
  v8bf lo = *(const v8bf*)(p);
  v8bf hi = *(const v8bf*)(p + 16);
  v16bf b;
#pragma unroll
  for (int i = 0; i < 8; ++i) { b[i] = lo[i]; b[i + 8] = hi[i]; }
  return b;
}

DEVINL v8f wmma_bf16(v16bf a, v16bf b, v8f c) {
  return __builtin_amdgcn_wmma_f32_16x16x32_bf16(false, a, false, b,
                                                 (short)0, c, false, false);
}

// --- 16-lane row-group max via DPP (pure VALU).
template <int CTRL>
DEVINL float dppf(float v) {
  int x = __builtin_amdgcn_update_dpp(0, __float_as_int(v), CTRL, 0xF, 0xF, true);
  return __int_as_float(x);
}
DEVINL float rowmax16_dpp(float v) {
  v = fmaxf(v, dppf<0xB1>(v));   // quad_perm [1,0,3,2]
  v = fmaxf(v, dppf<0x4E>(v));   // quad_perm [2,3,0,1]
  v = fmaxf(v, dppf<0x141>(v));  // row_half_mirror
  v = fmaxf(v, dppf<0x140>(v));  // row_mirror
  return v;
}

// --- gfx1250 async global->LDS copy (16B per lane, ASYNCcnt-tracked).
// VDST is the LDS byte address: low 32 bits of the shared-aperture flat addr.
DEVINL void async_copy_b128(bf16* lds_ptr, const bf16* gptr) {
  unsigned lds_off = (unsigned)(unsigned long long)lds_ptr;
  asm volatile("global_load_async_to_lds_b128 %0, %1, off"
               :: "v"(lds_off), "v"(gptr) : "memory");
}
DEVINL void wait_async0() {
  asm volatile("s_wait_asynccnt 0" ::: "memory");
}

// ---------------------------------------------------------------------------
// f32 -> bf16 conversion (n multiple of 4)
// ---------------------------------------------------------------------------
__global__ void f32_to_bf16_kernel(const float* __restrict__ in,
                                   bf16* __restrict__ out, int n4) {
  int i = blockIdx.x * blockDim.x + threadIdx.x;
  if (i < n4) {
    float4 f = ((const float4*)in)[i];
    v4bf o;
    o[0] = (bf16)f.x; o[1] = (bf16)f.y; o[2] = (bf16)f.z; o[3] = (bf16)f.w;
    ((v4bf*)out)[i] = o;
  }
}

// ---------------------------------------------------------------------------
// Projection GEMM:  C[4096,1024] = A_bf16[4096,1024] @ W_bf16[1024,1024]^T.
// Block = 8 waves -> 128x64 tile; 64x32 W strip staged in LDS per k-step.
// ---------------------------------------------------------------------------
enum { MODE_Q = 0, MODE_K = 1, MODE_V = 2, MODE_OUT = 3 };

template <int MODE>
__global__ void gemm16_kernel(const bf16* __restrict__ A,
                              const bf16* __restrict__ W,
                              const float* __restrict__ bias,
                              void* __restrict__ outp) {
  __shared__ __align__(16) bf16 Ws[64 * 32];    // [n][k] strip, ld = 32

  const int tid   = threadIdx.x;
  const int lane  = tid & 31;
  const int wave  = tid >> 5;
  const int mrow  = blockIdx.y * 128 + wave * 16;
  const int nbase = blockIdx.x * 64;
  const int K = NSTATE;

  const int wn = tid >> 2, wseg = tid & 3;      // 64 rows x 4 x 16B segments

  v8f acc[4] = {{}, {}, {}, {}};
  for (int kk = 0; kk < K; kk += 32) {
    *(v8bf*)(Ws + wn * 32 + wseg * 8) =
        *(const v8bf*)(W + (nbase + wn) * K + kk + wseg * 8);
    __syncthreads();

    v16bf a = load_frag_a(A + mrow * K + kk, K, lane);
#pragma unroll
    for (int t = 0; t < 4; ++t) {
      v16bf b = load_frag_b(Ws + (t * 16) * 32, 32, lane);
      acc[t] = wmma_bf16(a, b, acc[t]);
    }
    __syncthreads();
  }

  const int half = lane >> 4;
#pragma unroll
  for (int t = 0; t < 4; ++t) {
#pragma unroll
    for (int r = 0; r < 8; ++r) {
      const int row = mrow + r + 8 * half;            // token index [0,4096)
      const int col = nbase + t * 16 + (lane & 15);   // feature index [0,1024)
      float v = acc[t][r];
      if constexpr (MODE == MODE_Q)  v = (v + bias[col]) * QK_SCALE;
      if constexpr (MODE == MODE_K)  v = v * QK_SCALE;
      if constexpr (MODE == MODE_V)  v = v + bias[col];
      if constexpr (MODE == MODE_OUT) {
        ((float*)outp)[row * NSTATE + col] = v + bias[col];
      } else {
        const int bb = row >> 11, s = row & 2047;
        const int h  = col >> 6,  d = col & 63;
        bf16* o = (bf16*)outp;
        if constexpr (MODE == MODE_V)   // V^T : [B,H,Dh,S]
          o[(((bb << 4) + h) * 64 + d) * 2048 + s] = (bf16)v;
        else                            // Q,K : [B,H,S,Dh]
          o[(((bb << 4) + h) * 2048 + s) * 64 + d] = (bf16)v;
      }
    }
  }
}

// ---------------------------------------------------------------------------
// Flash attention.  8 waves/block cover 8 consecutive q-tiles of one (b,h).
// K/V blocks are double-buffered in LDS via async global->LDS DMA: iteration
// i computes from buf[i%2] while buf[(i+1)%2] is filled; one s_wait_asynccnt
// + one barrier per iteration.  Only the diagonal block takes the masked path.
// ---------------------------------------------------------------------------
__global__ void flash_attn_kernel(const bf16* __restrict__ Q,
                                  const bf16* __restrict__ Kt,
                                  const bf16* __restrict__ Vt,
                                  bf16* __restrict__ O) {
  __shared__ __align__(16) bf16 Ks[2][32 * 64];     // [key][d], ld=64 (2x4KB)
  __shared__ __align__(16) bf16 Vs[2][64 * 32];     // [d][key], ld=32 (2x4KB)
  __shared__ __align__(16) bf16 pstage[8][16 * 32]; // per-wave P tile  (8KB)

  const int tid  = threadIdx.x;
  const int lane = tid & 31;
  const int wave = tid >> 5;
  const int tile = blockIdx.x * 8 + wave;     // [0, 4096)
  const int bh   = tile >> 7;                 // (b*16 + h); same for all waves
  const int q0   = (tile & 127) * 16;
  const int bb   = bh >> 4, h = bh & 15;
  const int half = lane >> 4;
  const int qbase = ((blockIdx.x * 8) & 127) * 16;  // multiple of 128
  const int jend  = qbase + 128;              // causal range of last wave

  const bf16* Qbh = Q  + (size_t)bh * S * DHEAD;   // [S,Dh]
  const bf16* Kbh = Kt + (size_t)bh * S * DHEAD;   // [S,Dh]
  const bf16* Vbh = Vt + (size_t)bh * DHEAD * S;   // [Dh,S]

  // Q fragments for the whole tile (Dh=64 -> two K=32 fragments)
  v16bf aq0 = load_frag_a(Qbh + q0 * DHEAD + 0,  DHEAD, lane);
  v16bf aq1 = load_frag_a(Qbh + q0 * DHEAD + 32, DHEAD, lane);

  // all-ones B fragment: D = P x ones accumulates row sums of P
  v16bf ones;
#pragma unroll
  for (int i = 0; i < 16; ++i) ones[i] = (bf16)1.0f;

  v8f acc[4] = {{}, {}, {}, {}};
  v8f accl = {};                     // softmax denominator (replicated over n)
  float mrow[8];
#pragma unroll
  for (int r = 0; r < 8; ++r) mrow[r] = -3.0e38f;

  bf16* ps = &pstage[wave][0];
  const int krow = tid >> 3, kseg = tid & 7;  // K stage: 32 rows x 8 segs
  const int vrow = tid >> 2, vseg = tid & 3;  // V stage: 64 rows x 4 segs

  // async stage of K/V block j0 into buffer buf (16B per thread per tensor)
  auto stage = [&](int j0, int buf) {
    async_copy_b128(Ks[buf] + krow * 64 + kseg * 8,
                    Kbh + (j0 + krow) * DHEAD + kseg * 8);
    async_copy_b128(Vs[buf] + vrow * 32 + vseg * 8,
                    Vbh + vrow * S + j0 + vseg * 8);
  };

  // prologue: fill buffer 0
  stage(0, 0);
  wait_async0();
  __syncthreads();

  for (int j0 = 0; j0 < jend; j0 += 32) {
    const int cur = (j0 >> 5) & 1;
    if (j0 + 32 < jend) stage(j0 + 32, cur ^ 1);   // overlap next copy
    __builtin_prefetch(Kbh + (j0 + 64 + krow) * DHEAD + kseg * 8, 0, 1);

    if (j0 < q0 + 16) {                       // this wave's causal frontier
      const bf16* Kc = Ks[cur];
      const bf16* Vc = Vs[cur];
      // S tiles for two 16-col sub-blocks (Q 16x64 . K^T 64x32), K from LDS
      v8f s0 = {}, s1 = {};
      s0 = wmma_bf16(aq0, load_frag_b(Kc + 0 * 64 + 0,   64, lane), s0);
      s0 = wmma_bf16(aq1, load_frag_b(Kc + 0 * 64 + 32,  64, lane), s0);
      s1 = wmma_bf16(aq0, load_frag_b(Kc + 16 * 64 + 0,  64, lane), s1);
      s1 = wmma_bf16(aq1, load_frag_b(Kc + 16 * 64 + 32, 64, lane), s1);

      const bool diag = (j0 + 32 > q0);       // wave-uniform: scalar branch
      const int c0 = j0 + (lane & 15);
#pragma unroll
      for (int r = 0; r < 8; ++r) {
        float a0 = s0[r], a1 = s1[r];
        if (diag) {
          const int row = q0 + r + 8 * half;
          a0 += (c0 > row)      ? NEG_BIG : 0.0f;
          a1 += (c0 + 16 > row) ? NEG_BIG : 0.0f;
        }
        const float mx   = rowmax16_dpp(fmaxf(a0, a1));
        const float mnew = fmaxf(mrow[r], mx);
        const float corr = __expf(mrow[r] - mnew);
        mrow[r] = mnew;
        const float p0 = __expf(a0 - mnew);
        const float p1 = __expf(a1 - mnew);
        accl[r] *= corr;
#pragma unroll
        for (int t = 0; t < 4; ++t) acc[t][r] *= corr;
        ps[(r + 8 * half) * 32 +      (lane & 15)] = (bf16)p0;
        ps[(r + 8 * half) * 32 + 16 + (lane & 15)] = (bf16)p1;
      }
      // this wave's 16 ds_store_b16 must land before the transposed reload
      asm volatile("s_wait_dscnt 0" ::: "memory");
      v16bf ap = load_frag_a(ps, 32, lane);   // P tile 16x32 (A layout)
      accl = wmma_bf16(ap, ones, accl);       // denominator += rowsum(P)
#pragma unroll
      for (int t = 0; t < 4; ++t) {
        v16bf bv = load_frag_b(Vc + (t * 16) * 32, 32, lane); // V from LDS
        acc[t] = wmma_bf16(ap, bv, acc[t]);
      }
    }
    wait_async0();                            // next buffer fully in LDS
    __syncthreads();                          // everyone done with cur buffer
  }

  // epilogue: normalize and store O as [B,S,H*Dh] bf16 (k-contiguous rows)
#pragma unroll
  for (int r = 0; r < 8; ++r) {
    const float inv  = 1.0f / accl[r];
    const int   srow = q0 + r + 8 * half;
#pragma unroll
    for (int t = 0; t < 4; ++t) {
      const int col = h * DHEAD + t * 16 + (lane & 15);
      O[((size_t)bb * S + srow) * NSTATE + col] = (bf16)(acc[t][r] * inv);
    }
  }
}

// ---------------------------------------------------------------------------
// Host launcher.  Inputs: 0:x 1:mask 2:Wq 3:bq 4:Wk 5:Wv 6:bv 7:Wo 8:bo
// Workspace (48 MB): x_bf16 | Wq,Wk,Wv,Wo bf16 | Q | K | V^T | O  (all bf16)
// ---------------------------------------------------------------------------
extern "C" void kernel_launch(void* const* d_in, const int* in_sizes, int n_in,
                              void* d_out, int out_size, void* d_ws,
                              size_t ws_size, hipStream_t stream) {
  const float* x  = (const float*)d_in[0];
  const float* Wq = (const float*)d_in[2];
  const float* bq = (const float*)d_in[3];
  const float* Wk = (const float*)d_in[4];
  const float* Wv = (const float*)d_in[5];
  const float* bv = (const float*)d_in[6];
  const float* Wo = (const float*)d_in[7];
  const float* bo = (const float*)d_in[8];
  float* out = (float*)d_out;

  char* ws = (char*)d_ws;
  const size_t MB = 1u << 20;
  bf16* xb  = (bf16*)(ws + 0 * MB);    // 4096x1024        (8 MB)
  bf16* wqb = (bf16*)(ws + 8 * MB);    // 1024x1024        (2 MB)
  bf16* wkb = (bf16*)(ws + 10 * MB);
  bf16* wvb = (bf16*)(ws + 12 * MB);
  bf16* wob = (bf16*)(ws + 14 * MB);
  bf16* qws = (bf16*)(ws + 16 * MB);   // [B,H,S,Dh]       (8 MB)
  bf16* kws = (bf16*)(ws + 24 * MB);   // [B,H,S,Dh]
  bf16* vtw = (bf16*)(ws + 32 * MB);   // [B,H,Dh,S]
  bf16* ows = (bf16*)(ws + 40 * MB);   // [B,S,1024]

  // f32 -> bf16 conversions
  {
    const int nx4 = (Bn * S * NSTATE) / 4;
    f32_to_bf16_kernel<<<(nx4 + 255) / 256, 256, 0, stream>>>(x, xb, nx4);
    const int nw4 = (NSTATE * NSTATE) / 4;
    f32_to_bf16_kernel<<<(nw4 + 255) / 256, 256, 0, stream>>>(Wq, wqb, nw4);
    f32_to_bf16_kernel<<<(nw4 + 255) / 256, 256, 0, stream>>>(Wk, wkb, nw4);
    f32_to_bf16_kernel<<<(nw4 + 255) / 256, 256, 0, stream>>>(Wv, wvb, nw4);
    f32_to_bf16_kernel<<<(nw4 + 255) / 256, 256, 0, stream>>>(Wo, wob, nw4);
  }

  // projections: grid = (N/64, M/128), 256 threads (8 waves)
  dim3 gproj(NSTATE / 64, (Bn * S) / 128);
  gemm16_kernel<MODE_Q><<<gproj, 256, 0, stream>>>(xb, wqb, bq, qws);
  gemm16_kernel<MODE_K><<<gproj, 256, 0, stream>>>(xb, wkb, nullptr, kws);
  gemm16_kernel<MODE_V><<<gproj, 256, 0, stream>>>(xb, wvb, bv, vtw);

  // attention: 4096 query tiles / 8 waves per block, async K/V double-buffer
  flash_attn_kernel<<<512, 256, 0, stream>>>(qws, kws, vtw, ows);

  // output projection -> fp32 d_out
  gemm16_kernel<MODE_OUT><<<gproj, 256, 0, stream>>>(ows, wob, bo, out);
}
